// iRMB_25967372271807
// MI455X (gfx1250) — matrix operand order repaired and checked by reading, not compiled, not run on hardware
//
#include <hip/hip_runtime.h>

// ---------- types ----------
typedef __attribute__((ext_vector_type(16))) __bf16        v16bf;
typedef __attribute__((ext_vector_type(8)))  float         v8f;
typedef __attribute__((ext_vector_type(4)))  unsigned int  u32x4;

union BF16Frag { v16bf v; u32x4 q[2]; unsigned short s[16]; };
union F32x8    { v8f   v; float f[8]; };

__device__ __forceinline__ unsigned short f32_to_bf16(float f) {
    unsigned int u = __float_as_uint(f);
    u += 0x7FFFu + ((u >> 16) & 1u);          // round-to-nearest-even
    return (unsigned short)(u >> 16);
}
__device__ __forceinline__ v8f wmma_bf16(const v16bf& a, const v16bf& b, v8f c) {
    return __builtin_amdgcn_wmma_f32_16x16x32_bf16(false, a, false, b, (short)0, c, false, false);
}

// ---------- problem sizes ----------
#define BB    16
#define CC    256
#define HH    56
#define HWSZ  3136          // 56*56
#define NPIXT 50176         // B*H*W
#define WSZ   7
#define NPIX  49
#define NPAD  64
#define NHEAD 4
#define DHD   64
#define PITCH 52            // fp32 LDS row pitch (49 used)

// ---------- win_attn LDS layout (bytes) ----------
#define XW_OFF    0
#define XWB_OFF   53248                    // [64 pix][256 c] bf16 (dead after qkv GEMM)
#define S_OFF     53248                    // alias: S fp32 [64][64]
#define P_OFF     (53248 + 16384)          // alias: P bf16 [64][64]
#define QB_OFF    (53248 + 32768)          // q bf16 [4][64 pix][64 d]
#define KB_OFF    (QB_OFF + 32768)         // k bf16 [4][64 pix][64 d]
#define VB_OFF    (KB_OFF + 32768)         // v bf16 [256 c][64 pix]
#define VF_OFF    (VB_OFF + 32768)         // v fp32 [256][PITCH]
#define SPA_OFF   (VF_OFF + 53248)         // spa fp32 [256][PITCH]
#define SMEM_SZ   (SPA_OFF + 53248)        // 290816 B < 320 KB

// =====================================================================
// K0: BN batch statistics -> per-channel scale/shift
// =====================================================================
__global__ __launch_bounds__(256) void bn_stats_kernel(
    const float* __restrict__ x, const float* __restrict__ gamma,
    const float* __restrict__ beta, float* __restrict__ bn_a, float* __restrict__ bn_b)
{
    __shared__ float s_sum[256], s_sq[256];
    const int c = blockIdx.x, tid = threadIdx.x;
    float sum = 0.f, sq = 0.f;
    for (int i = tid; i < BB * HWSZ; i += 256) {
        int b = i / HWSZ, pos = i - b * HWSZ;
        float v = x[(b * CC + c) * HWSZ + pos];
        sum += v; sq += v * v;
    }
    s_sum[tid] = sum; s_sq[tid] = sq;
    __syncthreads();
    for (int off = 128; off > 0; off >>= 1) {
        if (tid < off) { s_sum[tid] += s_sum[tid + off]; s_sq[tid] += s_sq[tid + off]; }
        __syncthreads();
    }
    if (tid == 0) {
        const float N = (float)(BB * HWSZ);
        float mean = s_sum[0] / N;
        float var  = s_sq[0] / N - mean * mean;
        float a = gamma[c] * rsqrtf(var + 1e-5f);
        bn_a[c] = a;
        bn_b[c] = beta[c] - mean * a;
    }
}

// =====================================================================
// K1: fp32 -> bf16 weight conversion
// =====================================================================
__global__ void cvt_bf16_kernel(const float* __restrict__ src,
                                unsigned short* __restrict__ dst, int n)
{
    int i = blockIdx.x * blockDim.x + threadIdx.x;
    if (i < n) dst[i] = f32_to_bf16(src[i]);
}

// =====================================================================
// K2: per-window kernel: BN apply, qkv GEMM (WMMA), attention (WMMA),
//     3/5/7 depthwise branches, merge, write y (fp32) + y (bf16 pixel-major)
// =====================================================================
__global__ __launch_bounds__(256) void win_attn_kernel(
    const float* __restrict__ x,
    const float* __restrict__ bn_a, const float* __restrict__ bn_b,
    const unsigned short* __restrict__ Wqkv,   // [768][256] bf16
    const float* __restrict__ c3w, const float* __restrict__ c3b,
    const float* __restrict__ c5w, const float* __restrict__ c5b,
    const float* __restrict__ c7w, const float* __restrict__ c7b,
    float* __restrict__ y_out, unsigned short* __restrict__ y_bf16)
{
    extern __shared__ char smem[];
    float*          xw  = (float*)(smem + XW_OFF);              // [256][PITCH]
    unsigned short* xwb = (unsigned short*)(smem + XWB_OFF);    // [64][256]
    float*          Sm  = (float*)(smem + S_OFF);               // [64][64]
    unsigned short* Pm  = (unsigned short*)(smem + P_OFF);      // [64][64]
    unsigned short* qb  = (unsigned short*)(smem + QB_OFF);
    unsigned short* kb  = (unsigned short*)(smem + KB_OFF);
    unsigned short* vb  = (unsigned short*)(smem + VB_OFF);
    float*          vf  = (float*)(smem + VF_OFF);
    float*          spa = (float*)(smem + SPA_OFF);

    const int tid  = threadIdx.x;
    const int wave = tid >> 5;
    const int lane = tid & 31;
    const int ln   = lane & 15;
    const int lh   = lane >> 4;

    const int wid  = blockIdx.x;
    const int bidx = wid >> 6;
    const int n1   = (wid >> 3) & 7;
    const int n2   = wid & 7;

    // -------- Phase A: load + BN, fill xw (fp32) and xwb (bf16, zero-padded) ----
    for (int idx = tid; idx < CC * NPAD; idx += 256) {
        int c = idx >> 6, p = idx & 63;
        float val = 0.f;
        if (p < NPIX) {
            int h1 = p / WSZ, w1 = p - h1 * WSZ;
            int h = h1 * 8 + n1, w = w1 * 8 + n2;
            val = x[((bidx * CC + c) * HH + h) * HH + w] * bn_a[c] + bn_b[c];
            xw[c * PITCH + p] = val;
        }
        xwb[p * CC + c] = f32_to_bf16(val);
    }
    __syncthreads();

    // -------- Phase B: qkv GEMM, M=768 (48 tiles) x N=64 (4 tiles), K=256 ------
    // Class boundaries (q:0-255, k:256-511, v:512-767) are multiples of 16, so
    // the writeback branch is uniform per M-tile -> scalar branch on mt.
    for (int tile = wave; tile < 192; tile += 8) {
        const int mt = tile >> 2, nt = tile & 3;
        F32x8 acc;
        #pragma unroll
        for (int i = 0; i < 8; ++i) acc.f[i] = 0.f;
        const int row  = mt * 16 + ln;
        const int npix = nt * 16 + ln;
        const unsigned short* wrow = Wqkv + row * CC + lh * 8;
        for (int ks = 0; ks < 8; ++ks) {
            BF16Frag a, b;
            const unsigned short* ap = wrow + ks * 32;
            __builtin_prefetch(ap + 32, 0, 1);          // next K-step weight panel
            a.q[0] = *reinterpret_cast<const u32x4*>(ap);
            a.q[1] = *reinterpret_cast<const u32x4*>(ap + 16);
            const unsigned short* bp = xwb + npix * CC + ks * 32 + lh * 8;
            b.q[0] = *reinterpret_cast<const u32x4*>(bp);
            b.q[1] = *reinterpret_cast<const u32x4*>(bp + 16);
            acc.v = wmma_bf16(a.v, b.v, acc.v);
        }
        if (mt < 16) {                                  // Q -> [head][pix][d]
            #pragma unroll
            for (int r = 0; r < 8; ++r) {
                int m = mt * 16 + r + lh * 8;
                qb[(m >> 6) * 4096 + npix * DHD + (m & 63)] = f32_to_bf16(acc.f[r]);
            }
        } else if (mt < 32) {                           // K -> [head][pix][d]
            #pragma unroll
            for (int r = 0; r < 8; ++r) {
                int mm = mt * 16 + r + lh * 8 - 256;
                kb[(mm >> 6) * 4096 + npix * DHD + (mm & 63)] = f32_to_bf16(acc.f[r]);
            }
        } else {                                        // V (+ residual x) -> [c][pix]
            if (npix < NPIX) {
                #pragma unroll
                for (int r = 0; r < 8; ++r) {
                    int cc = mt * 16 + r + lh * 8 - 512;
                    float vv = acc.f[r] + xw[cc * PITCH + npix];
                    vf[cc * PITCH + npix] = vv;
                    vb[cc * NPAD + npix] = f32_to_bf16(vv);
                }
            } else {
                #pragma unroll
                for (int r = 0; r < 8; ++r) {
                    int cc = mt * 16 + r + lh * 8 - 512;
                    vb[cc * NPAD + npix] = 0;
                }
            }
        }
    }
    __syncthreads();

    // -------- Phase C: attention per head -------------------------------------
    const float scale = 0.125f;                  // 64^-0.5
    for (int hd = 0; hd < NHEAD; ++hd) {
        // S = q @ k^T  (M=64, N=64, K=64)
        for (int tile = wave; tile < 16; tile += 8) {
            const int it = tile >> 2, nt = tile & 3;
            F32x8 acc;
            #pragma unroll
            for (int i = 0; i < 8; ++i) acc.f[i] = 0.f;
            #pragma unroll
            for (int ks = 0; ks < 2; ++ks) {
                BF16Frag a, b;
                const unsigned short* ap = qb + hd * 4096 + (it * 16 + ln) * DHD + ks * 32 + lh * 8;
                a.q[0] = *reinterpret_cast<const u32x4*>(ap);
                a.q[1] = *reinterpret_cast<const u32x4*>(ap + 16);
                const unsigned short* bp = kb + hd * 4096 + (nt * 16 + ln) * DHD + ks * 32 + lh * 8;
                b.q[0] = *reinterpret_cast<const u32x4*>(bp);
                b.q[1] = *reinterpret_cast<const u32x4*>(bp + 16);
                acc.v = wmma_bf16(a.v, b.v, acc.v);
            }
            #pragma unroll
            for (int r = 0; r < 8; ++r) {
                int m = it * 16 + r + lh * 8;
                Sm[m * NPAD + nt * 16 + ln] = acc.f[r] * scale;
            }
        }
        __syncthreads();

        // softmax (rows 0..48), pad rows/cols -> 0
        for (int r = tid; r < NPAD; r += 256) {
            if (r < NPIX) {
                float mx = -1e30f;
                for (int j = 0; j < NPIX; ++j) mx = fmaxf(mx, Sm[r * NPAD + j]);
                float sum = 0.f;
                for (int j = 0; j < NPIX; ++j) sum += __expf(Sm[r * NPAD + j] - mx);
                float inv = 1.f / sum;
                for (int j = 0; j < NPAD; ++j)
                    Pm[r * NPAD + j] = (j < NPIX) ? f32_to_bf16(__expf(Sm[r * NPAD + j] - mx) * inv)
                                                  : (unsigned short)0;
            } else {
                for (int j = 0; j < NPAD; ++j) Pm[r * NPAD + j] = 0;
            }
        }
        __syncthreads();

        // out = P @ V  (M=64 pix, N=64 d, K=64 pix)
        for (int tile = wave; tile < 16; tile += 8) {
            const int it = tile >> 2, dt = tile & 3;
            F32x8 acc;
            #pragma unroll
            for (int i = 0; i < 8; ++i) acc.f[i] = 0.f;
            #pragma unroll
            for (int ks = 0; ks < 2; ++ks) {
                BF16Frag a, b;
                const unsigned short* ap = Pm + (it * 16 + ln) * NPAD + ks * 32 + lh * 8;
                a.q[0] = *reinterpret_cast<const u32x4*>(ap);
                a.q[1] = *reinterpret_cast<const u32x4*>(ap + 16);
                const unsigned short* bp = vb + (hd * DHD + dt * 16 + ln) * NPAD + ks * 32 + lh * 8;
                b.q[0] = *reinterpret_cast<const u32x4*>(bp);
                b.q[1] = *reinterpret_cast<const u32x4*>(bp + 16);
                acc.v = wmma_bf16(a.v, b.v, acc.v);
            }
            const int ch = hd * DHD + dt * 16 + ln;
            #pragma unroll
            for (int r = 0; r < 8; ++r) {
                int m = it * 16 + r + lh * 8;       // query pixel
                if (m < NPIX) spa[ch * PITCH + m] = acc.f[r];
            }
        }
        __syncthreads();
    }

    // -------- Phase D: depthwise 3/5/7 branches, blend, +V residual, merge ----
    for (int idx = tid; idx < CC * NPIX; idx += 256) {
        const int c = idx / NPIX, p = idx - c * NPIX;
        const int h1 = p / WSZ, w1 = p - h1 * WSZ;

        float a3 = c3b[c];
        for (int dy = -1; dy <= 1; ++dy)
            for (int dx = -1; dx <= 1; ++dx) {
                int yy = h1 + dy, xx = w1 + dx;
                if ((unsigned)yy < WSZ && (unsigned)xx < WSZ)
                    a3 += xw[c * PITCH + yy * WSZ + xx] * c3w[c * 9 + (dy + 1) * 3 + (dx + 1)];
            }
        float a5 = c5b[c];
        for (int dy = -2; dy <= 2; ++dy)
            for (int dx = -2; dx <= 2; ++dx) {
                int yy = h1 + dy, xx = w1 + dx;
                if ((unsigned)yy < WSZ && (unsigned)xx < WSZ)
                    a5 += xw[c * PITCH + yy * WSZ + xx] * c5w[c * 25 + (dy + 2) * 5 + (dx + 2)];
            }
        float a7 = c7b[c];
        for (int dy = -3; dy <= 3; ++dy)
            for (int dx = -3; dx <= 3; ++dx) {
                int yy = h1 + dy, xx = w1 + dx;
                if ((unsigned)yy < WSZ && (unsigned)xx < WSZ)
                    a7 += xw[c * PITCH + yy * WSZ + xx] * c7w[c * 49 + (dy + 3) * 7 + (dx + 3)];
            }

        float yv = 0.25f * (spa[c * PITCH + p] + a3 + a5 + a7) + vf[c * PITCH + p];
        const int h = h1 * 8 + n1, w = w1 * 8 + n2;
        y_out[((bidx * CC + c) * HH + h) * HH + w] = yv;
        y_bf16[((bidx * HH + h) * HH + w) * CC + c] = f32_to_bf16(yv);
    }
}

// =====================================================================
// K3: ffn_in 1x1 GEMM (256x256, N=64 pixels/WG) -> t (fp32 NCHW) + bias
// =====================================================================
__global__ __launch_bounds__(256) void ffn_in_kernel(
    const unsigned short* __restrict__ Win,    // [256][256] bf16
    const float* __restrict__ bias,
    const unsigned short* __restrict__ yb,     // [50176][256] bf16 pixel-major
    float* __restrict__ t)
{
    const int wave = threadIdx.x >> 5, lane = threadIdx.x & 31;
    const int ln = lane & 15, lh = lane >> 4;
    const long base = (long)blockIdx.x * 64;

    for (int tile = wave; tile < 64; tile += 8) {
        const int mt = tile >> 2, nt = tile & 3;
        F32x8 acc;
        #pragma unroll
        for (int i = 0; i < 8; ++i) acc.f[i] = 0.f;
        const int  row = mt * 16 + ln;
        const long pix = base + nt * 16 + ln;
        const unsigned short* wrow = Win + row * CC + lh * 8;
        const unsigned short* brow = yb + pix * CC + lh * 8;
        #pragma unroll
        for (int ks = 0; ks < 8; ++ks) {
            BF16Frag a, b;
            const unsigned short* ap = wrow + ks * 32;
            __builtin_prefetch(ap + 32, 0, 1);
            a.q[0] = *reinterpret_cast<const u32x4*>(ap);
            a.q[1] = *reinterpret_cast<const u32x4*>(ap + 16);
            const unsigned short* bp = brow + ks * 32;
            b.q[0] = *reinterpret_cast<const u32x4*>(bp);
            b.q[1] = *reinterpret_cast<const u32x4*>(bp + 16);
            acc.v = wmma_bf16(a.v, b.v, acc.v);
        }
        const int gp  = (int)pix;
        const int bb  = gp / HWSZ;
        const int rem = gp - bb * HWSZ;
        #pragma unroll
        for (int r = 0; r < 8; ++r) {
            int m = mt * 16 + r + lh * 8;
            t[(bb * CC + m) * HWSZ + rem] = acc.f[r] + bias[m];
        }
    }
}

// =====================================================================
// K4: dw3x3(t)+SiLU -> s (bf16 in LDS), ffn_out GEMM + bias + shortcut
// =====================================================================
__global__ __launch_bounds__(256) void ffn_out_kernel(
    const float* __restrict__ t,
    const float* __restrict__ clw, const float* __restrict__ clb,
    const unsigned short* __restrict__ Wout, const float* __restrict__ bias,
    const float* __restrict__ ysc, float* __restrict__ out)
{
    __shared__ unsigned short s_lds[64 * CC];     // 32 KB: [pix][c]

    const int wg = blockIdx.x;
    const int b  = wg / 49;
    const int tl = wg - b * 49;
    const int h0 = (tl / 7) * 8, w0 = (tl % 7) * 8;

    for (int idx = threadIdx.x; idx < CC * 64; idx += 256) {
        const int c = idx >> 6, p = idx & 63;
        const int h = h0 + (p >> 3), w = w0 + (p & 7);
        float acc = clb[c];
        for (int dy = -1; dy <= 1; ++dy)
            for (int dx = -1; dx <= 1; ++dx) {
                int yy = h + dy, xx = w + dx;
                if ((unsigned)yy < HH && (unsigned)xx < HH)
                    acc += t[(b * CC + c) * HWSZ + yy * HH + xx] * clw[c * 9 + (dy + 1) * 3 + (dx + 1)];
            }
        float s = acc / (1.f + __expf(-acc));     // SiLU
        s_lds[p * CC + c] = f32_to_bf16(s);
    }
    __syncthreads();

    const int wave = threadIdx.x >> 5, lane = threadIdx.x & 31;
    const int ln = lane & 15, lh = lane >> 4;
    for (int tile = wave; tile < 64; tile += 8) {
        const int mt = tile >> 2, nt = tile & 3;
        F32x8 acc;
        #pragma unroll
        for (int i = 0; i < 8; ++i) acc.f[i] = 0.f;
        const int row = mt * 16 + ln;
        const int p   = nt * 16 + ln;
        const unsigned short* wrow = Wout + row * CC + lh * 8;
        #pragma unroll
        for (int ks = 0; ks < 8; ++ks) {
            BF16Frag a, bfr;
            const unsigned short* ap = wrow + ks * 32;
            __builtin_prefetch(ap + 32, 0, 1);
            a.q[0] = *reinterpret_cast<const u32x4*>(ap);
            a.q[1] = *reinterpret_cast<const u32x4*>(ap + 16);
            const unsigned short* bp = s_lds + p * CC + ks * 32 + lh * 8;
            bfr.q[0] = *reinterpret_cast<const u32x4*>(bp);
            bfr.q[1] = *reinterpret_cast<const u32x4*>(bp + 16);
            acc.v = wmma_bf16(a.v, bfr.v, acc.v);
        }
        const int h = h0 + (p >> 3), w = w0 + (p & 7);
        #pragma unroll
        for (int r = 0; r < 8; ++r) {
            int m = mt * 16 + r + lh * 8;
            long gi = (long)(b * CC + m) * HWSZ + h * HH + w;
            out[gi] = acc.f[r] + bias[m] + ysc[gi];
        }
    }
}

// =====================================================================
// Launch
// =====================================================================
extern "C" void kernel_launch(void* const* d_in, const int* in_sizes, int n_in,
                              void* d_out, int out_size, void* d_ws, size_t ws_size,
                              hipStream_t stream)
{
    (void)in_sizes; (void)n_in; (void)out_size; (void)ws_size;

    const float* x        = (const float*)d_in[0];
    const float* bn_gamma = (const float*)d_in[1];
    const float* bn_beta  = (const float*)d_in[2];
    const float* qkv_w    = (const float*)d_in[3];
    const float* c3w      = (const float*)d_in[4];
    const float* c3b      = (const float*)d_in[5];
    const float* c5w      = (const float*)d_in[6];
    const float* c5b      = (const float*)d_in[7];
    const float* c7w      = (const float*)d_in[8];
    const float* c7b      = (const float*)d_in[9];
    const float* fin_w    = (const float*)d_in[10];
    const float* fin_b    = (const float*)d_in[11];
    const float* clw      = (const float*)d_in[12];
    const float* clb      = (const float*)d_in[13];
    const float* fout_w   = (const float*)d_in[14];
    const float* fout_b   = (const float*)d_in[15];

    char* ws = (char*)d_ws;
    float*          bn_a   = (float*)(ws + 0);
    float*          bn_b   = (float*)(ws + 1024);
    unsigned short* Wqkv   = (unsigned short*)(ws + 2048);          // 768*256
    unsigned short* Win    = (unsigned short*)(ws + 395264);        // 256*256
    unsigned short* Wout   = (unsigned short*)(ws + 526336);        // 256*256
    float*          y_f32  = (float*)(ws + 657408);                 // B*C*H*W
    unsigned short* y_bf16 = (unsigned short*)(ws + 52037632);      // pix-major
    float*          t_f32  = (float*)(ws + 77727744);               // B*C*H*W

    bn_stats_kernel<<<CC, 256, 0, stream>>>(x, bn_gamma, bn_beta, bn_a, bn_b);

    cvt_bf16_kernel<<<(768 * 256 + 255) / 256, 256, 0, stream>>>(qkv_w, Wqkv, 768 * 256);
    cvt_bf16_kernel<<<(256 * 256 + 255) / 256, 256, 0, stream>>>(fin_w, Win, 256 * 256);
    cvt_bf16_kernel<<<(256 * 256 + 255) / 256, 256, 0, stream>>>(fout_w, Wout, 256 * 256);

    (void)hipFuncSetAttribute((const void*)win_attn_kernel,
                              hipFuncAttributeMaxDynamicSharedMemorySize, SMEM_SZ);
    win_attn_kernel<<<1024, 256, SMEM_SZ, stream>>>(
        x, bn_a, bn_b, Wqkv, c3w, c3b, c5w, c5b, c7w, c7b, y_f32, y_bf16);

    ffn_in_kernel<<<NPIXT / 64, 256, 0, stream>>>(Win, fin_b, y_bf16, t_f32);

    ffn_out_kernel<<<BB * 49, 256, 0, stream>>>(
        t_f32, clw, clb, Wout, fout_b, y_f32, (float*)d_out);
}
